// TA_GCN3D_unit_k3d_32409823216278
// MI455X (gfx1250) — matrix operand order repaired and checked by reading, not compiled, not run on hardware
//
#include <hip/hip_runtime.h>

#define Nn 128
#define Cc 64
#define Tt 128
#define Vv 25
#define Ss 3
#define Hh 8
#define IH 8
#define Dd 200
#define DPAD 224
#define TOT 26214400   // N*C*T*V
#define TCK 8          // t-chunks for gcn-attention partial accumulation

typedef __attribute__((ext_vector_type(16))) _Float16 v16h;
typedef __attribute__((ext_vector_type(8)))  float    v8f;

__device__ inline v8f wmma_f16(v16h a, v16h b, v8f c) {
  return __builtin_amdgcn_wmma_f32_16x16x32_f16(false, a, false, b, (short)0, c, false, false);
}

// wave32 16-lane xor reductions via ds_swizzle (group-of-32, and=0x1f)
__device__ inline float xor16_max(float v) {
  v = fmaxf(v, __int_as_float(__builtin_amdgcn_ds_swizzle(__float_as_int(v), 0x041F)));
  v = fmaxf(v, __int_as_float(__builtin_amdgcn_ds_swizzle(__float_as_int(v), 0x081F)));
  v = fmaxf(v, __int_as_float(__builtin_amdgcn_ds_swizzle(__float_as_int(v), 0x101F)));
  v = fmaxf(v, __int_as_float(__builtin_amdgcn_ds_swizzle(__float_as_int(v), 0x201F)));
  return v;
}
__device__ inline float xor16_sum(float v) {
  v += __int_as_float(__builtin_amdgcn_ds_swizzle(__float_as_int(v), 0x041F));
  v += __int_as_float(__builtin_amdgcn_ds_swizzle(__float_as_int(v), 0x081F));
  v += __int_as_float(__builtin_amdgcn_ds_swizzle(__float_as_int(v), 0x101F));
  v += __int_as_float(__builtin_amdgcn_ds_swizzle(__float_as_int(v), 0x201F));
  return v;
}
__device__ inline float sigm(float x) { return 1.f / (1.f + __expf(-x)); }

// ---------------- 1a. GCN adaptive attention, partial sums over a 16-step t-chunk --------
// att_part[n, ck, s, v, w] = sum_{t in chunk} sum_i q[i,t,v] k[i,t,w]
__global__ __launch_bounds__(256) void k_gcnatt_part(const float* __restrict__ x,
    const float* __restrict__ Wq, const float* __restrict__ bq,
    const float* __restrict__ Wk, const float* __restrict__ bk,
    float* __restrict__ attp) {
  __shared__ float xsh[Cc * Vv];      // 64x25
  __shared__ float qk[800];           // q(16x25) | k(16x25)
  __shared__ float att[Ss * 625];
  const int b = blockIdx.x, n = b / TCK, ck = b % TCK;
  const int t0 = ck * (Tt / TCK);
  const int tid = threadIdx.x;
  for (int e = tid; e < Ss * 625; e += 256) att[e] = 0.f;
  __syncthreads();
  for (int tt = 0; tt < Tt / TCK; ++tt) {
    const int t = t0 + tt;
    for (int e = tid; e < Cc * Vv; e += 256)
      xsh[e] = x[(((size_t)n * Cc + e / Vv) * Tt + t) * Vv + (e % Vv)];
    __syncthreads();
    for (int s = 0; s < Ss; ++s) {
      for (int e = tid; e < 800; e += 256) {
        const int isK = (e >= 400);
        const int e2 = isK ? e - 400 : e;
        const int i = e2 / Vv, v = e2 % Vv;
        const float* W = (isK ? Wk : Wq) + s * 16 * Cc;
        float acc = (isK ? bk : bq)[s * 16 + i];
        for (int c = 0; c < Cc; ++c) acc += W[i * Cc + c] * xsh[c * Vv + v];
        qk[e] = acc;
      }
      __syncthreads();
      for (int e = tid; e < 625; e += 256) {
        const int v = e / Vv, w = e % Vv;
        float acc = 0.f;
        for (int i = 0; i < 16; ++i) acc += qk[i * Vv + v] * qk[400 + i * Vv + w];
        att[s * 625 + e] += acc;
      }
      __syncthreads();
    }
  }
  for (int e = tid; e < Ss * 625; e += 256)
    attp[((size_t)n * TCK + ck) * (Ss * 625) + e] = att[e];
}

// ---------------- 1b. reduce partials, softmax over v (dim -2), Aad = A + alpha*p --------
__global__ __launch_bounds__(256) void k_gcnatt_fin(const float* __restrict__ attp,
    const float* __restrict__ A, const float* __restrict__ alpha_p,
    float* __restrict__ aad) {
  __shared__ float att[Ss * 625];
  __shared__ float mxs[75], sms[75];
  const int n = blockIdx.x, tid = threadIdx.x;
  for (int e = tid; e < Ss * 625; e += 256) {
    float s = 0.f;
    for (int ck = 0; ck < TCK; ++ck)
      s += attp[((size_t)n * TCK + ck) * (Ss * 625) + e];
    att[e] = s;
  }
  __syncthreads();
  const float inv = 1.f / 2048.f;   // 1/(ig*T)
  if (tid < 75) {
    const int s = tid / Vv, w = tid % Vv;
    float m = -3.0e38f;
    for (int v = 0; v < Vv; ++v) m = fmaxf(m, att[s * 625 + v * Vv + w]);
    float sum = 0.f;
    for (int v = 0; v < Vv; ++v) sum += __expf((att[s * 625 + v * Vv + w] - m) * inv);
    mxs[tid] = m; sms[tid] = sum;
  }
  __syncthreads();
  const float alpha = alpha_p[0];
  for (int e = tid; e < Ss * 625; e += 256) {
    const int s = e / 625, r = e % 625, w = r % Vv;
    const float p = __expf((att[e] - mxs[s * Vv + w]) * inv) / sms[s * Vv + w];
    aad[(size_t)n * (Ss * 625) + e] = A[e] + alpha * p;
  }
}

// ---------------- 2. GCN aggregation + Wf GEMM (WMMA) + BN + res + ReLU ----------------
__global__ __launch_bounds__(256) void k_gcn(const float* __restrict__ x,
    const float* __restrict__ aad, const float* __restrict__ Wf,
    const float* __restrict__ bf, const float* __restrict__ g_gamma,
    const float* __restrict__ g_beta, float* __restrict__ y0) {
  __shared__ float     Al[Ss * 625];          // 7500 B
  __shared__ _Float16  xs[Cc * 50];           // 6400 B
  __shared__ _Float16  Wl[Cc * 192];          // 24576 B
  __shared__ _Float16  aggB[192 * 64];        // 24576 B
  const int b = blockIdx.x, n = b >> 6, t0 = (b & 63) * 2;
  const int tid = threadIdx.x, w = tid >> 5, lane = tid & 31;
  const int m16 = lane & 15, g = lane >> 4;
  for (int e = tid; e < Ss * 625; e += 256) Al[e] = aad[(size_t)n * (Ss * 625) + e];
  for (int e = tid; e < Cc * 50; e += 256) {
    const int c = e / 50, col = e % 50, ti = col / Vv, v = col % Vv;
    xs[e] = (_Float16)x[(((size_t)n * Cc + c) * Tt + t0 + ti) * Vv + v];
  }
  for (int e = tid; e < Cc * 192; e += 256) {
    const int o = e / 192, k = e % 192, s = k >> 6, c = k & 63;
    Wl[e] = (_Float16)Wf[(s * Cc + o) * Cc + c];
  }
  __syncthreads();
  for (int e = tid; e < 192 * 64; e += 256) {
    const int k = e >> 6, col = e & 63;
    float acc = 0.f;
    if (col < 50) {
      const int s = k >> 6, c = k & 63, ti = col / Vv, wv = col % Vv;
      const float* Ad = Al + s * 625;
      for (int v = 0; v < Vv; ++v) acc += (float)xs[c * 50 + ti * Vv + v] * Ad[v * Vv + wv];
    }
    aggB[e] = (_Float16)acc;
  }
  __syncthreads();
  const float sE = rsqrtf(1.f + 1e-5f);
  for (int cb = w; cb < 16; cb += 8) {
    const int mt = cb >> 2, nt = cb & 3;
    v8f acc = {};
#pragma unroll
    for (int kc = 0; kc < 6; ++kc) {
      v16h aa, bb;
#pragma unroll
      for (int e = 0; e < 16; ++e)
        aa[e] = Wl[(mt * 16 + m16) * 192 + kc * 32 + 16 * g + e];
#pragma unroll
      for (int e = 0; e < 16; ++e)
        bb[e] = aggB[(kc * 32 + 16 * g + e) * 64 + nt * 16 + m16];
      acc = wmma_f16(aa, bb, acc);
    }
    const int col = nt * 16 + m16;
    if (col < 50) {
      const int ti = col / Vv, vv = col % Vv;
#pragma unroll
      for (int r = 0; r < 8; ++r) {
        const int o = mt * 16 + r + 8 * g;
        const float bfs = bf[o] + bf[64 + o] + bf[128 + o];
        const size_t idx = (((size_t)n * Cc + o) * Tt + t0 + ti) * Vv + vv;
        float val = acc[r] + bfs;
        val = val * (g_gamma[o] * sE) + g_beta[o] + x[idx];
        y0[idx] = fmaxf(val, 0.f);
      }
    }
  }
}

// ---------------- 3-5. attention gates ----------------
__global__ __launch_bounds__(256) void k_g1(const float* __restrict__ y0,
    const float* __restrict__ W_sa, const float* __restrict__ b_sa, float* __restrict__ sa) {
  __shared__ float se1[Cc * Vv];
  const int n = blockIdx.x, tid = threadIdx.x;
  for (int e = tid; e < Cc * Vv; e += 256) {
    const int c = e / Vv, v = e % Vv;
    const float* p = y0 + (((size_t)n * Cc + c) * Tt) * Vv + v;
    float s = 0.f;
    for (int t = 0; t < Tt; ++t) {
      __builtin_prefetch(&p[(t + 24) * Vv], 0, 1);   // global_prefetch_b8 ahead of stride-100B walk
      s += p[t * Vv];
    }
    se1[e] = s * (1.f / Tt);
  }
  __syncthreads();
  if (tid < Vv) {
    float acc = b_sa[0];
    for (int c = 0; c < Cc; ++c)
      for (int j = 0; j < Vv; ++j) {
        const int idx = tid + j - 12;
        if (idx >= 0 && idx < Vv) acc += W_sa[c * Vv + j] * se1[c * Vv + idx];
      }
    sa[n * Vv + tid] = sigm(acc);
  }
}

__global__ __launch_bounds__(256) void k_g2(const float* __restrict__ y0,
    const float* __restrict__ sa, const float* __restrict__ W_ta,
    const float* __restrict__ b_ta, float* __restrict__ ta, float* __restrict__ se2g) {
  __shared__ float se2[Cc * Tt];
  const int n = blockIdx.x, tid = threadIdx.x;
  for (int e = tid; e < Cc * Tt; e += 256) {
    const int c = e / Tt, t = e % Tt;
    const float* p = y0 + (((size_t)n * Cc + c) * Tt + t) * Vv;
    __builtin_prefetch(&p[256 * Vv], 0, 1);
    float s = 0.f;
    for (int v = 0; v < Vv; ++v) s += p[v] * (1.f + sa[n * Vv + v]);
    const float m = s * (1.f / Vv);
    se2[e] = m; se2g[(size_t)n * Cc * Tt + e] = m;
  }
  __syncthreads();
  if (tid < Tt) {
    float acc = b_ta[0];
    for (int c = 0; c < Cc; ++c)
      for (int j = 0; j < 9; ++j) {
        const int idx = tid + j - 4;
        if (idx >= 0 && idx < Tt) acc += W_ta[c * 9 + j] * se2[c * Tt + idx];
      }
    ta[n * Tt + tid] = sigm(acc);
  }
}

__global__ __launch_bounds__(64) void k_g3(const float* __restrict__ se2g,
    const float* __restrict__ ta, const float* __restrict__ W_fc1,
    const float* __restrict__ b_fc1, const float* __restrict__ W_fc2,
    const float* __restrict__ b_fc2, float* __restrict__ ca) {
  __shared__ float se3[Cc];
  __shared__ float hid[32];
  const int n = blockIdx.x, tid = threadIdx.x;
  {
    float s = 0.f;
    const float* p = se2g + (size_t)n * Cc * Tt + tid * Tt;
    __builtin_prefetch(p, 0, 1);
    for (int t = 0; t < Tt; ++t) s += p[t] * (1.f + ta[n * Tt + t]);
    se3[tid] = s * (1.f / Tt);
  }
  __syncthreads();
  if (tid < 32) {
    float a = b_fc1[tid];
    for (int c = 0; c < Cc; ++c) a += W_fc1[tid * Cc + c] * se3[c];
    hid[tid] = fmaxf(a, 0.f);
  }
  __syncthreads();
  {
    float a = b_fc2[tid];
    for (int j = 0; j < 32; ++j) a += W_fc2[tid * 32 + j] * hid[j];
    ca[n * Cc + tid] = sigm(a);
  }
}

// ---------------- 6. apply gates + pos-emb ----------------
__global__ __launch_bounds__(256) void k_xa(const float* __restrict__ y0,
    const float* __restrict__ sa, const float* __restrict__ ta,
    const float* __restrict__ ca, const float* __restrict__ pe, float* __restrict__ xa) {
  const size_t i = (size_t)blockIdx.x * 256 + threadIdx.x;
  if (i >= (size_t)TOT) return;
  const int v = (int)(i % Vv);
  size_t r = i / Vv;
  const int t = (int)(r % Tt); r /= Tt;
  const int c = (int)(r % Cc);
  const int n = (int)(r / Cc);
  xa[i] = y0[i] * (1.f + sa[n * Vv + v]) * (1.f + ta[n * Tt + t]) * (1.f + ca[n * Cc + c])
        + pe[(t * Cc + c) * Vv + v];
}

// ---------------- 7. fused QKV projection (WMMA, 192x64 x 64xcols) ----------------
__global__ __launch_bounds__(256) void k_qkv(const float* __restrict__ xa,
    const float* __restrict__ Wqh, const float* __restrict__ bqh,
    const float* __restrict__ Wkh, const float* __restrict__ bkh,
    const float* __restrict__ Wvh, const float* __restrict__ bvh,
    float* __restrict__ qh, float* __restrict__ kh, float* __restrict__ vh) {
  __shared__ _Float16 Wl[192 * 64];
  __shared__ _Float16 Bt[64 * 128];
  const int b = blockIdx.x, n = b / 25, col0 = (b % 25) * 128;
  const int tid = threadIdx.x, w = tid >> 5, lane = tid & 31;
  const int m16 = lane & 15, g = lane >> 4;
  for (int e = tid; e < 192 * 64; e += 256) {
    const int og = e >> 6, c = e & 63;
    const float v = (og < 64) ? Wqh[og * 64 + c]
                  : (og < 128) ? Wkh[(og - 64) * 64 + c] : Wvh[(og - 128) * 64 + c];
    Wl[e] = (_Float16)v;
  }
  for (int e = tid; e < 64 * 128; e += 256) {
    const int c = e >> 7, j = e & 127, col = col0 + j, t = col / Vv, v = col % Vv;
    Bt[e] = (_Float16)xa[(((size_t)n * Cc + c) * Tt + t) * Vv + v];
  }
  __syncthreads();
  for (int mt = 0; mt < 12; ++mt) {
    v8f acc = {};
#pragma unroll
    for (int kc = 0; kc < 2; ++kc) {
      v16h aa, bb;
#pragma unroll
      for (int e = 0; e < 16; ++e)
        aa[e] = Wl[(mt * 16 + m16) * 64 + kc * 32 + 16 * g + e];
#pragma unroll
      for (int e = 0; e < 16; ++e)
        bb[e] = Bt[(kc * 32 + 16 * g + e) * 128 + w * 16 + m16];
      acc = wmma_f16(aa, bb, acc);
    }
    const int col = col0 + w * 16 + m16, t = col / Vv, vv = col % Vv;
#pragma unroll
    for (int r = 0; r < 8; ++r) {
      const int og = mt * 16 + r + 8 * g;
      const int proj = og >> 6, oc = og & 63, h = oc >> 3;
      const size_t dst = (((size_t)n * Hh + h) * Tt + t) * Dd + (oc & 7) * Vv + vv;
      const float bias = (proj == 0 ? bqh : proj == 1 ? bkh : bvh)[oc];
      float* outp = (proj == 0 ? qh : proj == 1 ? kh : vh);
      outp[dst] = acc[r] + bias;
    }
  }
}

// ---------------- 8. causal MHA per (n,h): S=QK^T, softmax, O=PV (WMMA) + bn + xa ----------------
__global__ __launch_bounds__(256) void k_attn(const float* __restrict__ qh,
    const float* __restrict__ kh, const float* __restrict__ vh,
    const float* __restrict__ xa, const float* __restrict__ m_gamma,
    const float* __restrict__ m_beta, float* __restrict__ zbuf) {
  __shared__ _Float16 smem[DPAD * Tt];  // 57344 B; phase2 reuses as P(16384)+V(2048)
  const int b = blockIdx.x, n = b / Hh, h = b % Hh;
  const int tid = threadIdx.x, w = tid >> 5, lane = tid & 31;
  const int m16 = lane & 15, g = lane >> 4;
  const int trow_base = 16 * w;
  const float qscale = 0.0707106781f;   // 1/sqrt(200)
  const float* kbase = kh + ((size_t)(n * Hh + h)) * Tt * Dd;
  const float* qbase = qh + ((size_t)(n * Hh + h)) * Tt * Dd;
  const float* vbase = vh + ((size_t)(n * Hh + h)) * Tt * Dd;
  // stage K^T (d-major) in LDS, zero-pad d to 224
  for (int idx = tid; idx < DPAD * Tt; idx += 256) {
    const int d = idx / Tt, s = idx % Tt;
    smem[idx] = (_Float16)((d < Dd) ? kbase[s * Dd + d] : 0.f);
  }
  // Q row-tile fragments (scaled)
  v16h aq[7];
#pragma unroll
  for (int kc = 0; kc < 7; ++kc)
#pragma unroll
    for (int e = 0; e < 16; ++e) {
      const int d = kc * 32 + 16 * g + e;
      aq[kc][e] = (_Float16)((d < Dd) ? qbase[(trow_base + m16) * Dd + d] * qscale : 0.f);
    }
  __syncthreads();
  v8f S[8];
#pragma unroll
  for (int j = 0; j < 8; ++j) {
    v8f acc = {};
#pragma unroll
    for (int kc = 0; kc < 7; ++kc) {
      v16h bb;
#pragma unroll
      for (int e = 0; e < 16; ++e)
        bb[e] = smem[(kc * 32 + 16 * g + e) * Tt + 16 * j + m16];
      acc = wmma_f16(aq[kc], bb, acc);
    }
    S[j] = acc;
  }
  // causal mask + row softmax (rows spread over 16-lane halves + 8 tiles)
#pragma unroll
  for (int r = 0; r < 8; ++r) {
    const int trow = trow_base + r + 8 * g;
    float mx = -3.0e38f;
#pragma unroll
    for (int j = 0; j < 8; ++j) {
      const int scol = 16 * j + m16;
      const float val = (scol <= trow) ? S[j][r] : -1.0e9f;
      S[j][r] = val;
      mx = fmaxf(mx, val);
    }
    mx = xor16_max(mx);
    float sum = 0.f;
#pragma unroll
    for (int j = 0; j < 8; ++j) { const float p = __expf(S[j][r] - mx); S[j][r] = p; sum += p; }
    sum = xor16_sum(sum);
    const float inv = 1.f / sum;
#pragma unroll
    for (int j = 0; j < 8; ++j) S[j][r] *= inv;
  }
  __syncthreads();   // everyone done reading K region
  _Float16* Psh = smem;                 // 8 * 16 * 128
  _Float16* Vsh = smem + 8 * 16 * Tt;   // 128 * 16
#pragma unroll
  for (int r = 0; r < 8; ++r)
#pragma unroll
    for (int j = 0; j < 8; ++j)
      Psh[w * (16 * Tt) + (r + 8 * g) * Tt + 16 * j + m16] = (_Float16)S[j][r];
  __syncthreads();
  const float sE = rsqrtf(1.f + 1e-5f);
  for (int dc = 0; dc < 13; ++dc) {
    for (int idx = tid; idx < Tt * 16; idx += 256) {
      const int s = idx >> 4, cl = idx & 15, d = dc * 16 + cl;
      Vsh[idx] = (_Float16)((d < Dd) ? vbase[s * Dd + d] : 0.f);
    }
    __syncthreads();
    v8f acc = {};
#pragma unroll
    for (int kc = 0; kc < 4; ++kc) {
      v16h aa, bb;
#pragma unroll
      for (int e = 0; e < 16; ++e)
        aa[e] = Psh[w * (16 * Tt) + m16 * Tt + kc * 32 + 16 * g + e];
#pragma unroll
      for (int e = 0; e < 16; ++e)
        bb[e] = Vsh[(kc * 32 + 16 * g + e) * 16 + m16];
      acc = wmma_f16(aa, bb, acc);
    }
    const int dout = dc * 16 + m16;
    if (dout < Dd) {
      const int c = h * IH + dout / Vv, vv = dout % Vv;
      const float sc = m_gamma[c] * sE, sb = m_beta[c];
#pragma unroll
      for (int r = 0; r < 8; ++r) {
        const int t = trow_base + r + 8 * g;
        const size_t idx = (((size_t)n * Cc + c) * Tt + t) * Vv + vv;
        zbuf[idx] = acc[r] * sc + sb + xa[idx];   // z1 = bn(attn) + xa
      }
    }
    __syncthreads();
  }
}

// ---------------- 9. FFN GEMM (WMMA) + relu + xa + bn + final relu(z+x) ----------------
__global__ __launch_bounds__(256) void k_ffn(const float* __restrict__ zbuf,
    const float* __restrict__ xa, const float* __restrict__ x,
    const float* __restrict__ Wffn, const float* __restrict__ bffn,
    const float* __restrict__ m_gamma, const float* __restrict__ m_beta,
    float* __restrict__ out) {
  __shared__ _Float16 Wl[64 * 64];
  __shared__ _Float16 Bt[64 * 128];
  const int b = blockIdx.x, n = b / 25, col0 = (b % 25) * 128;
  const int tid = threadIdx.x, w = tid >> 5, lane = tid & 31;
  const int m16 = lane & 15, g = lane >> 4;
  for (int e = tid; e < 64 * 64; e += 256) Wl[e] = (_Float16)Wffn[e];
  for (int e = tid; e < 64 * 128; e += 256) {
    const int c = e >> 7, j = e & 127, col = col0 + j, t = col / Vv, v = col % Vv;
    Bt[e] = (_Float16)zbuf[(((size_t)n * Cc + c) * Tt + t) * Vv + v];
  }
  __syncthreads();
  const float sE = rsqrtf(1.f + 1e-5f);
  for (int mt = 0; mt < 4; ++mt) {
    v8f acc = {};
#pragma unroll
    for (int kc = 0; kc < 2; ++kc) {
      v16h aa, bb;
#pragma unroll
      for (int e = 0; e < 16; ++e)
        aa[e] = Wl[(mt * 16 + m16) * 64 + kc * 32 + 16 * g + e];
#pragma unroll
      for (int e = 0; e < 16; ++e)
        bb[e] = Bt[(kc * 32 + 16 * g + e) * 128 + w * 16 + m16];
      acc = wmma_f16(aa, bb, acc);
    }
    const int col = col0 + w * 16 + m16, t = col / Vv, vv = col % Vv;
#pragma unroll
    for (int r = 0; r < 8; ++r) {
      const int o = mt * 16 + r + 8 * g;
      const size_t idx = (((size_t)n * Cc + o) * Tt + t) * Vv + vv;
      float val = fmaxf(acc[r] + bffn[o], 0.f) + xa[idx];
      val = val * (m_gamma[o] * sE) + m_beta[o];
      out[idx] = fmaxf(val + x[idx], 0.f);
    }
  }
}

extern "C" void kernel_launch(void* const* d_in, const int* in_sizes, int n_in,
                              void* d_out, int out_size, void* d_ws, size_t ws_size,
                              hipStream_t stream) {
  const float* x       = (const float*)d_in[0];
  const float* A       = (const float*)d_in[1];
  const float* alpha   = (const float*)d_in[2];
  const float* Wq      = (const float*)d_in[3];
  const float* bq      = (const float*)d_in[4];
  const float* Wk      = (const float*)d_in[5];
  const float* bk      = (const float*)d_in[6];
  const float* Wf      = (const float*)d_in[7];
  const float* bf      = (const float*)d_in[8];
  const float* g_gamma = (const float*)d_in[9];
  const float* g_beta  = (const float*)d_in[10];
  const float* W_sa    = (const float*)d_in[11];
  const float* b_sa    = (const float*)d_in[12];
  const float* W_ta    = (const float*)d_in[13];
  const float* b_ta    = (const float*)d_in[14];
  const float* W_fc1   = (const float*)d_in[15];
  const float* b_fc1   = (const float*)d_in[16];
  const float* W_fc2   = (const float*)d_in[17];
  const float* b_fc2   = (const float*)d_in[18];
  const float* pe      = (const float*)d_in[19];
  const float* Wqh     = (const float*)d_in[20];
  const float* bqh     = (const float*)d_in[21];
  const float* Wkh     = (const float*)d_in[22];
  const float* bkh     = (const float*)d_in[23];
  const float* Wvh     = (const float*)d_in[24];
  const float* bvh     = (const float*)d_in[25];
  const float* Wffn    = (const float*)d_in[26];
  const float* bffn    = (const float*)d_in[27];
  const float* m_gamma = (const float*)d_in[28];
  const float* m_beta  = (const float*)d_in[29];

  float* ws   = (float*)d_ws;
  float* AAD  = ws;                          // 240000
  float* Y0   = AAD + 240000;                // 26214400 (reused as z1 buffer later)
  float* XA   = Y0 + (size_t)TOT;            // 26214400
  float* QH   = XA + (size_t)TOT;            // 26214400
  float* KH   = QH + (size_t)TOT;            // 26214400
  float* VH   = KH + (size_t)TOT;            // 26214400
  float* SAb  = VH + (size_t)TOT;            // 3200
  float* TAb  = SAb + Nn * Vv;               // 16384
  float* CAb  = TAb + Nn * Tt;               // 8192
  float* SE2  = CAb + Nn * Cc;               // 1048576
  float* ATTP = SE2 + (size_t)Nn * Cc * Tt;  // 1920000  (N * TCK * S * 625)
  float* ZB   = Y0;                          // alias: Y0 dead after k_xa

  k_gcnatt_part<<<Nn * TCK, 256, 0, stream>>>(x, Wq, bq, Wk, bk, ATTP);
  k_gcnatt_fin<<<Nn, 256, 0, stream>>>(ATTP, A, alpha, AAD);
  k_gcn<<<Nn * (Tt / 2), 256, 0, stream>>>(x, AAD, Wf, bf, g_gamma, g_beta, Y0);
  k_g1<<<Nn, 256, 0, stream>>>(Y0, W_sa, b_sa, SAb);
  k_g2<<<Nn, 256, 0, stream>>>(Y0, SAb, W_ta, b_ta, TAb, SE2);
  k_g3<<<Nn, 64, 0, stream>>>(SE2, TAb, W_fc1, b_fc1, W_fc2, b_fc2, CAb);
  k_xa<<<(TOT + 255) / 256, 256, 0, stream>>>(Y0, SAb, TAb, CAb, pe, XA);
  k_qkv<<<Nn * 25, 256, 0, stream>>>(XA, Wqh, bqh, Wkh, bkh, Wvh, bvh, QH, KH, VH);
  k_attn<<<Nn * Hh, 256, 0, stream>>>(QH, KH, VH, XA, m_gamma, m_beta, ZB);
  k_ffn<<<Nn * 25, 256, 0, stream>>>(ZB, XA, x, Wffn, bffn, m_gamma, m_beta, (float*)d_out);
}
